// RegionPurposalNetwork_33775622815749
// MI455X (gfx1250) — matrix operand order repaired and checked by reading, not compiled, not run on hardware
//
#include <hip/hip_runtime.h>
#include <hip/hip_bf16.h>
#include <cmath>

// CDNA5 wave32 WMMA fragment types
typedef __attribute__((ext_vector_type(16))) _Float16 v16h;
typedef __attribute__((ext_vector_type(8)))  float    v8f;

// obj > 0.95  <=>  (l0 - l1) > ln(0.95/0.05) = ln(19)
#define LOGIT_CONF 2.94443897916644046f
#define TILES_PER_WAVE 4

// ---------------------------------------------------------------------------
// order-preserving float -> u32 encode (handles negatives), and decode
// ---------------------------------------------------------------------------
__device__ inline unsigned enc_f32(float f) {
  unsigned u = __float_as_uint(f);
  unsigned mask = (u & 0x80000000u) ? 0xFFFFFFFFu : 0x80000000u;
  return u ^ mask;
}
__device__ inline float dec_f32(unsigned e) {
  unsigned u = (e & 0x80000000u) ? (e ^ 0x80000000u) : ~e;
  return __uint_as_float(u);
}

// ---------------------------------------------------------------------------
// Kernel A: conv1  X(1,H,W) -> Y(H*W) interleaved float2 (ch0, ch1), f32.
// 2D grid: no integer division anywhere. Also resets the argmax cell.
// ---------------------------------------------------------------------------
__global__ void rpn_conv1_kernel(const float* __restrict__ X,
                                 const float* __restrict__ conv_w,  // (2,1,3,3)
                                 const float* __restrict__ conv_b,  // (2,)
                                 float2* __restrict__ Y,
                                 unsigned long long* __restrict__ best,
                                 int W, int H) {
  const int x = blockIdx.x * blockDim.x + threadIdx.x;
  const int y = blockIdx.y;
  if (blockIdx.x == 0 && blockIdx.y == 0 && threadIdx.x == 0) *best = 0ULL;
  if (x >= W) return;

  // stream-ahead prefetch of the row below (global_prefetch_b8)
  if (y + 1 < H) __builtin_prefetch(&X[(long long)(y + 1) * W + x], 0, 1);

  float acc0 = conv_b[0];
  float acc1 = conv_b[1];
#pragma unroll
  for (int i = 0; i < 3; ++i) {
#pragma unroll
    for (int j = 0; j < 3; ++j) {
      const int yy = y + i - 1, xx = x + j - 1;
      const bool in = (yy >= 0) && (yy < H) && (xx >= 0) && (xx < W);
      const float v = in ? X[(long long)yy * W + xx] : 0.0f;
      acc0 = fmaf(conv_w[i * 3 + j],     v, acc0);
      acc1 = fmaf(conv_w[9 + i * 3 + j], v, acc1);
    }
  }
  Y[(long long)y * W + x] = make_float2(acc0, acc1);
}

// ---------------------------------------------------------------------------
// Kernel B: cls conv per 16-pixel tile via v_wmma_f32_16x16x32_f16; rank by
// t = l0 - l1 (monotone with softmax ch0) -> packed-u64 argmax reduction.
// 2D grid (tiles-in-row x rows). Interior tiles (96.7%) take a uniform
// fast path with unconditional loads (no exec manipulation); frame tiles
// take the guarded path. Both are wave-uniform -> EXEC all-ones at WMMA.
// ---------------------------------------------------------------------------
__device__ inline unsigned long long shflxor_u64(unsigned long long v, int m) {
  unsigned lo = (unsigned)v, hi = (unsigned)(v >> 32);
  lo = __shfl_xor(lo, m, 32);
  hi = __shfl_xor(hi, m, 32);
  return ((unsigned long long)hi << 32) | lo;
}

__global__ void rpn_cls_argmax_kernel(const float2* __restrict__ Y,
                                      const float* __restrict__ cls_w,  // (2,2,3,3)
                                      const float* __restrict__ cls_b,  // (2,)
                                      unsigned long long* __restrict__ best,
                                      int W, int H) {
  const int lane = threadIdx.x & 31;
  const bool lo  = lane < 16;
  const int  M   = lane & 15;
  // wave index forced scalar so all tile math stays in SALU
  const int waveB = __builtin_amdgcn_readfirstlane((int)(threadIdx.x >> 5));
  const int tilesPerRow = W >> 4;                  // W % 16 == 0
  const int y  = blockIdx.y;                       // pixel row
  const int tileInRow0 = (blockIdx.x * 8 + waveB) * TILES_PER_WAVE;
  const bool interiorY = (y > 0) && (y < H - 1);   // wave-uniform

  // ---- A fragment: 16x32 f16, rows 0..1 = cls weights (k = c*9+i*3+j) ----
  // lane<16 : elems 0..15 -> K0..15
  // lane>=16: elems 0..15 -> K16..31
  v16h a;
#pragma unroll
  for (int e = 0; e < 16; ++e) {
    const int k = (lo ? 0 : 16) + e;
    const float wv = (M < 2 && k < 18) ? cls_w[M * 18 + k] : 0.0f;
    a[e] = (_Float16)wv;
  }
  const float bdiff = cls_b[0] - cls_b[1];

  unsigned long long key = 0ULL;

#pragma unroll
  for (int tt = 0; tt < TILES_PER_WAVE; ++tt) {
    const int tileInRow = tileInRow0 + tt;
    if (tileInRow >= tilesPerRow) break;           // wave-uniform
    const int x = (tileInRow << 4) + M;
    const int p = y * W + x;

    // ---- B fragment source: per-pixel 2ch x 3x3 neighborhood ----
    float2 t[9];
    const bool interior = interiorY &&
                          (tileInRow > 0) && (tileInRow < tilesPerRow - 1);
    if (interior) {                                // uniform fast path
      const float2* r0 = Y + ((long long)(y - 1) * W + (x - 1));
      const float2* r1 = r0 + W;
      const float2* r2 = r1 + W;
      t[0] = r0[0]; t[1] = r0[1]; t[2] = r0[2];
      t[3] = r1[0]; t[4] = r1[1]; t[5] = r1[2];
      t[6] = r2[0]; t[7] = r2[1]; t[8] = r2[2];
    } else {                                       // frame tiles: guarded
#pragma unroll
      for (int i = 0; i < 3; ++i) {
#pragma unroll
        for (int j = 0; j < 3; ++j) {
          const int tap = i * 3 + j;
          const int yy = y + i - 1, xx = x + j - 1;
          const bool need = lo || (tap >= 7);
          const bool in = need && (yy >= 0) && (yy < H) &&
                                  (xx >= 0) && (xx < W);
          t[tap] = in ? Y[(long long)yy * W + xx] : make_float2(0.0f, 0.0f);
        }
      }
    }

    // lane<16 : elems 0..15 -> K0..15 ; lane>=16 : elems 0..15 -> K16..31
    const _Float16 z = (_Float16)0.0f;
    v16h b;
    b[0]  = lo ? (_Float16)t[0].x : (_Float16)t[7].y;   // K0  | K16
    b[1]  = lo ? (_Float16)t[1].x : (_Float16)t[8].y;   // K1  | K17
    b[2]  = lo ? (_Float16)t[2].x : z;
    b[3]  = lo ? (_Float16)t[3].x : z;
    b[4]  = lo ? (_Float16)t[4].x : z;
    b[5]  = lo ? (_Float16)t[5].x : z;
    b[6]  = lo ? (_Float16)t[6].x : z;
    b[7]  = lo ? (_Float16)t[7].x : z;
    b[8]  = lo ? (_Float16)t[8].x : z;                  // K8  | 0
    b[9]  = lo ? (_Float16)t[0].y : z;                  // K9  | 0
    b[10] = lo ? (_Float16)t[1].y : z;
    b[11] = lo ? (_Float16)t[2].y : z;
    b[12] = lo ? (_Float16)t[3].y : z;
    b[13] = lo ? (_Float16)t[4].y : z;
    b[14] = lo ? (_Float16)t[5].y : z;
    b[15] = lo ? (_Float16)t[6].y : z;                  // K15 | 0

    v8f c = {};
    c = __builtin_amdgcn_wmma_f32_16x16x32_f16(
        /*neg_a=*/false, a, /*neg_b=*/false, b,
        /*c_mod=*/(short)0, c, /*reuse_a=*/false, /*reuse_b=*/false);

    // C layout: VGPR r, lanes 0..15 -> M=r, N=lane. Rows 0/1 = logits.
    if (lo) {
      const float tval = (c[0] - c[1]) + bdiff;         // monotone with obj
      const unsigned long long k =
          ((unsigned long long)enc_f32(tval) << 32) |
          (unsigned)(~(unsigned)p);                     // tie -> lowest index
      key = (k > key) ? k : key;
    }
  }

  // wave32 butterfly max-reduce, then block reduce, one atomic per block
#pragma unroll
  for (int o = 16; o > 0; o >>= 1) {
    const unsigned long long other = shflxor_u64(key, o);
    key = (other > key) ? other : key;
  }
  __shared__ unsigned long long sk[8];
  if (lane == 0) sk[waveB] = key;
  __syncthreads();
  if (threadIdx.x == 0) {
    unsigned long long m = sk[0];
#pragma unroll
    for (int i = 1; i < 8; ++i) m = (sk[i] > m) ? sk[i] : m;
    atomicMax(best, m);
  }
}

// ---------------------------------------------------------------------------
// Kernel C: winner pixel only. 20 threads = 5 anchors x 4 coords. Full f32:
// bbox conv channel (a*4+d), anchor decode (np.round == RNE), clip, store.
// ---------------------------------------------------------------------------
__global__ void rpn_finalize_kernel(const float2* __restrict__ Y,
                                    const float* __restrict__ bbox_w,  // (36,2,3,3)
                                    const float* __restrict__ bbox_b,  // (36,)
                                    const unsigned long long* __restrict__ best,
                                    float* __restrict__ out,
                                    int W, int H) {
  const unsigned long long key = *best;
  const float tmax = dec_f32((unsigned)(key >> 32));
  int p = (int)(~(unsigned)key);
  if (!(tmax > LOGIT_CONF)) p = 0;     // nothing above threshold -> idx 0..4

  const int tid = threadIdx.x;
  if (tid >= 20) return;
  const int x = p % W, y = p / W;

  float off = bbox_b[tid];
#pragma unroll
  for (int i = 0; i < 3; ++i) {
#pragma unroll
    for (int j = 0; j < 3; ++j) {
      const int yy = y + i - 1, xx = x + j - 1;
      const bool in = (yy >= 0) && (yy < H) && (xx >= 0) && (xx < W);
      const float2 v = in ? Y[(long long)yy * W + xx] : make_float2(0.0f, 0.0f);
      off = fmaf(bbox_w[tid * 18 + i * 3 + j],     v.x, off);
      off = fmaf(bbox_w[tid * 18 + 9 + i * 3 + j], v.y, off);
    }
  }

  const int a = tid >> 2, d = tid & 3;       // anchor, coord
  const int r = a / 3, s = a % 3;            // ratio-major, size-minor
  const float RA[3] = {0.5f, 1.0f, 2.0f};
  const float SZ[3] = {16.0f, 32.0f, 64.0f};
  const float hr = sqrtf(RA[r]);
  const float wr = 1.0f / hr;
  const float wsz = wr * SZ[s], hsz = hr * SZ[s];
  const float halfext = (d == 0) ? -0.5f * wsz
                      : (d == 1) ? -0.5f * hsz
                      : (d == 2) ?  0.5f * wsz
                                 :  0.5f * hsz;
  const float base  = rintf(halfext);                       // np.round (RNE)
  const float shift = (d & 1) ? (float)y : (float)x;
  const float lim   = (d & 1) ? (float)H : (float)W;
  float box = base + shift + off;
  box = fminf(fmaxf(box, 0.0f), lim);
  out[tid] = box;
}

// ---------------------------------------------------------------------------
extern "C" void kernel_launch(void* const* d_in, const int* in_sizes, int n_in,
                              void* d_out, int out_size, void* d_ws, size_t ws_size,
                              hipStream_t stream) {
  const float* X      = (const float*)d_in[0];
  const float* conv_w = (const float*)d_in[1];
  const float* conv_b = (const float*)d_in[2];
  const float* bbox_w = (const float*)d_in[3];
  const float* bbox_b = (const float*)d_in[4];
  const float* cls_w  = (const float*)d_in[5];
  const float* cls_b  = (const float*)d_in[6];

  const long long HW = in_sizes[0];                 // X is (1,H,W), H == W
  const int W = (int)llround(sqrt((double)HW));
  const int H = (int)(HW / W);

  // ws layout: [0..7] argmax u64 cell, [16..] Y as float2[H*W] (~8 MB)
  unsigned long long* best = (unsigned long long*)d_ws;
  float2* Y = (float2*)((char*)d_ws + 16);

  dim3 gridA((W + 255) / 256, H);
  rpn_conv1_kernel<<<gridA, 256, 0, stream>>>(X, conv_w, conv_b, Y, best, W, H);

  const int tilesPerRow = W >> 4;                   // W % 16 == 0
  const int wavesPerRow = (tilesPerRow + TILES_PER_WAVE - 1) / TILES_PER_WAVE;
  dim3 gridB((wavesPerRow + 7) / 8, H);             // 8 waves per block
  rpn_cls_argmax_kernel<<<gridB, 256, 0, stream>>>(Y, cls_w, cls_b, best, W, H);

  rpn_finalize_kernel<<<1, 32, 0, stream>>>(Y, bbox_w, bbox_b, best,
                                            (float*)d_out, W, H);
}
